// OverKillerRNN_40956808134895
// MI455X (gfx1250) — matrix operand order repaired and checked by reading, not compiled, not run on hardware
//
#include <hip/hip_runtime.h>
#include <hip/hip_bf16.h>

typedef __attribute__((ext_vector_type(16))) _Float16 v16h;
typedef __attribute__((ext_vector_type(8)))  float    v8f;
typedef __attribute__((ext_vector_type(4)))  int      v4i;

union F16x16 { v16h h; v4i i[2]; };

#define VOCAB 8000
#define DIM   128
#define UHID  256
#define BATCH 32
#define SEQ   256
#define G3    768          // 3*U
#define NPAD  8064         // 63*128
#define NT_W  (NPAD/16)    // 504

static __device__ __forceinline__ v8f wmma_f16(v16h a, v16h b, v8f c) {
    return __builtin_amdgcn_wmma_f32_16x16x32_f16(false, a, false, b, (short)0, c, false, false);
}

// -------------------------------------------------------------------------
// 1) Embedding gather + x-projection (both directions), fp32 out.
//    xz layout: [dir][s*32+b][768]
// -------------------------------------------------------------------------
__global__ __launch_bounds__(256) void embed_xproj(
    const int* __restrict__ tokens, const float* __restrict__ emb,
    const float* __restrict__ Wf, const float* __restrict__ Wb,
    const float* __restrict__ bf, const float* __restrict__ bb,
    float* __restrict__ xz)
{
    __shared__ float xl[8][DIM];
    const int t  = threadIdx.x;
    const int f0 = blockIdx.x * 8;            // flat token index s*32+b
    {   // stage 8 embedding rows into LDS
        int ti = t >> 5, d0 = (t & 31) * 4;
        int f = f0 + ti, s = f >> 5, b = f & 31;
        int tok = tokens[b * SEQ + s];
        const float* e = emb + (size_t)tok * DIM + d0;
        #pragma unroll
        for (int q = 0; q < 4; ++q) xl[ti][d0 + q] = e[q];
    }
    __syncthreads();
    #pragma unroll
    for (int rep = 0; rep < 6; ++rep) {
        int o   = t + rep * 256;              // 0..1535
        int dir = o / G3;
        int j   = o % G3;
        const float* W  = dir ? Wb : Wf;
        const float* b0 = dir ? bb : bf;      // row 0 of [2][768]
        float acc[8];
        #pragma unroll
        for (int ti = 0; ti < 8; ++ti) acc[ti] = 0.f;
        for (int d = 0; d < DIM; ++d) {
            float w = W[(size_t)d * G3 + j];
            #pragma unroll
            for (int ti = 0; ti < 8; ++ti) acc[ti] += xl[ti][d] * w;
        }
        float bias = b0[j];
        float* dst = xz + (size_t)dir * 8192 * G3 + (size_t)f0 * G3 + j;
        #pragma unroll
        for (int ti = 0; ti < 8; ++ti) dst[(size_t)ti * G3] = acc[ti] + bias;
    }
}

// -------------------------------------------------------------------------
// 2) Pack fp32 row-major [K][N] weight into WMMA B-fragment layout (f16):
//    dst[((nt*KC + kc)*512) + L*16 + i] = W[kc*32 + (L&16)/2 + i + (i&8)][nt*16 + L%16]
//    zero-fill columns >= N (padding).
// -------------------------------------------------------------------------
__global__ __launch_bounds__(256) void pack_w(
    const float* __restrict__ src, _Float16* __restrict__ dst,
    int KC, int N, long total)
{
    long g = (long)blockIdx.x * 256 + threadIdx.x;
    if (g >= total) return;
    int i  = (int)(g & 15);
    int L  = (int)((g >> 4) & 31);
    int kc = (int)((g >> 9) % KC);
    int nt = (int)(g / ((long)KC << 9));
    int k = kc * 32 + ((L & 16) >> 1) + i + (i & 8);
    int n = nt * 16 + (L & 15);
    float v = (n < N) ? src[(size_t)k * N + n] : 0.f;
    dst[g] = (_Float16)v;
}

// -------------------------------------------------------------------------
// 3) GRU scan. One block per direction, 256 threads = 8 waves
//    (2 waves/SIMD -> full VGPR budget, no spills).
//    Wave w owns u-column groups {2w, 2w+1}, BOTH batch halves (m0=0,16),
//    and all three gates -> 12 accumulator tiles, gates combined in regs.
//    h kept as f16 in LDS, padded row stride 272 halves.
//    B (packed U) re-streamed from L2 each step; an opaque integer offset
//    (empty asm) inside the step loop blocks loop-invariant hoisting while
//    keeping addrspace(1) inference -> global_load_b128, split waits.
// -------------------------------------------------------------------------
#define HPAD 272
__global__ __launch_bounds__(256) void gru_scan(
    const float* __restrict__ xz, const _Float16* __restrict__ Upk,
    const float* __restrict__ bf, const float* __restrict__ bb,
    const int* __restrict__ tokens, _Float16* __restrict__ ys)
{
    __shared__ _Float16 h16[32 * HPAD];
    const int dir = blockIdx.x;
    const float*    xzd = xz  + (size_t)dir * 8192 * G3;
    const _Float16* Ud  = Upk + (size_t)dir * 196608;
    const float*    b1  = (dir ? bb : bf) + G3;      // b[1]
    const int t = threadIdx.x;
    const int L = t & 31;
    const int w = t >> 5;                            // 0..7
    const int lhalf = (L < 16) ? 0 : 8;              // A k-offset
    const int bhalf = (L < 16) ? 0 : 8;              // C/D row offset

    for (int idx = t; idx < 32 * HPAD; idx += 256) h16[idx] = (_Float16)0.f;

    // bias preload: 2 u-groups x 3 gates
    float bg[2][3];
    #pragma unroll
    for (int ngi = 0; ngi < 2; ++ngi) {
        int u = (2 * w + ngi) * 16 + (L & 15);
        bg[ngi][0] = b1[u];
        bg[ngi][1] = b1[256 + u];
        bg[ngi][2] = b1[512 + u];
    }
    __syncthreads();

    for (int step = 0; step < SEQ; ++step) {
        const int s = dir ? (SEQ - 1 - step) : step;
        v8f acc[2][2][3];                            // [ngi][mi][gate]
        #pragma unroll
        for (int ngi = 0; ngi < 2; ++ngi)
            #pragma unroll
            for (int mi = 0; mi < 2; ++mi)
                #pragma unroll
                for (int g = 0; g < 3; ++g) acc[ngi][mi][g] = (v8f){};

        // opaque per-step offset: forbids hoisting the (loop-invariant)
        // B loads out of the time loop without breaking the global-pointer
        // address-space inference (keeps global_load_b128, not flat/spill).
        size_t ustep_off = 0;
        asm volatile("" : "+s"(ustep_off));
        const _Float16* Ustep = Ud + ustep_off;

        #pragma unroll
        for (int kc = 0; kc < 8; ++kc) {
            F16x16 A[2];                             // both batch halves
            #pragma unroll
            for (int mi = 0; mi < 2; ++mi) {
                const int abase = (mi * 16 + (L & 15)) * HPAD + kc * 32 + lhalf;
                A[mi].i[0] = *(const v4i*)&h16[abase];
                A[mi].i[1] = *(const v4i*)&h16[abase + 16];
            }
            #pragma unroll
            for (int ngi = 0; ngi < 2; ++ngi)
                #pragma unroll
                for (int g = 0; g < 3; ++g) {
                    const int nt = (2 * w + ngi) + 16 * g;
                    const _Float16* bp = Ustep + (((size_t)(nt * 8 + kc)) << 9) + (L << 4);
                    F16x16 B;
                    B.i[0] = *(const v4i*)bp;
                    B.i[1] = *(const v4i*)(bp + 8);
                    acc[ngi][0][g] = wmma_f16(A[0].h, B.h, acc[ngi][0][g]);
                    acc[ngi][1][g] = wmma_f16(A[1].h, B.h, acc[ngi][1][g]);
                }
        }
        __syncthreads();   // all A reads of h16 complete
        #pragma unroll
        for (int ngi = 0; ngi < 2; ++ngi) {
            const int u = (2 * w + ngi) * 16 + (L & 15);
            #pragma unroll
            for (int mi = 0; mi < 2; ++mi)
                #pragma unroll
                for (int r = 0; r < 8; ++r) {
                    const int b = mi * 16 + r + bhalf;
                    const size_t xrow = ((size_t)(s * 32 + b)) * G3;
                    const float iz = acc[ngi][mi][0][r] + bg[ngi][0];
                    const float ir = acc[ngi][mi][1][r] + bg[ngi][1];
                    const float ih = acc[ngi][mi][2][r] + bg[ngi][2];
                    const float xg = xzd[xrow + u];
                    const float xr = xzd[xrow + 256 + u];
                    const float xh = xzd[xrow + 512 + u];
                    const float z  = 1.f / (1.f + __expf(-(xg + iz)));
                    const float rr = 1.f / (1.f + __expf(-(xr + ir)));
                    const float hh = tanhf(xh + rr * ih);
                    const float hold = (float)h16[b * HPAD + u];
                    float hnew = z * hold + (1.f - z) * hh;
                    if (tokens[b * SEQ + s] == 0) hnew = hold;   // masked: carry
                    h16[b * HPAD + u] = (_Float16)hnew;
                    ys[((size_t)b * SEQ + s) * 512 + dir * 256 + u] = (_Float16)hnew;
                }
        }
        __syncthreads();   // h writes visible before next step's A reads
    }
}

// -------------------------------------------------------------------------
// 4) WMMA GEMM: out[M x Nout] = A(f16,[M x K]) @ Bpacked + bias.
//    Block tile 128x128, 8 waves (2x4), wave tile 64x32.
//    A via LDS (padded stride 40 halves); B fragments direct from packed
//    global buffer (L2-resident).
// -------------------------------------------------------------------------
#define ALDS_STRIDE 40
__global__ __launch_bounds__(256) void gemm_f16(
    const _Float16* __restrict__ A, int K, int KC,
    const _Float16* __restrict__ Bp, const float* __restrict__ bias,
    void* __restrict__ out, int out_f16, int Nout)
{
    __shared__ _Float16 Al[128 * ALDS_STRIDE];
    const int t = threadIdx.x;
    const int L = t & 31;
    const int w = t >> 5;
    const int wm = w >> 2;           // 0..1
    const int wn = w & 3;            // 0..3
    const int m0 = blockIdx.y * 128;
    const int n0 = blockIdx.x * 128;
    const int lhalf = (L < 16) ? 0 : 8;

    v8f acc[4][2];
    #pragma unroll
    for (int i = 0; i < 4; ++i)
        #pragma unroll
        for (int j = 0; j < 2; ++j) acc[i][j] = (v8f){};

    const int ldrow = t >> 1, ldseg = t & 1;
    for (int kc = 0; kc < KC; ++kc) {
        {   // stage A tile [128 x 32] into LDS
            const v4i* g = (const v4i*)(A + (size_t)(m0 + ldrow) * K + kc * 32 + ldseg * 16);
            v4i x0 = g[0], x1 = g[1];
            v4i* d = (v4i*)&Al[ldrow * ALDS_STRIDE + ldseg * 16];
            d[0] = x0; d[1] = x1;
        }
        __syncthreads();
        F16x16 Bfr[2];
        #pragma unroll
        for (int j = 0; j < 2; ++j) {
            const int nt = (n0 >> 4) + wn * 2 + j;
            const _Float16* bp = Bp + ((size_t)(nt * KC + kc) << 9) + (L << 4);
            Bfr[j].i[0] = *(const v4i*)bp;
            Bfr[j].i[1] = *(const v4i*)(bp + 8);
        }
        #pragma unroll
        for (int i = 0; i < 4; ++i) {
            const int rowb = wm * 64 + i * 16 + (L & 15);
            const int ab = rowb * ALDS_STRIDE + lhalf;
            F16x16 Af;
            Af.i[0] = *(const v4i*)&Al[ab];
            Af.i[1] = *(const v4i*)&Al[ab + 16];
            #pragma unroll
            for (int j = 0; j < 2; ++j)
                acc[i][j] = wmma_f16(Af.h, Bfr[j].h, acc[i][j]);
        }
        __syncthreads();
    }
    // epilogue: C/D layout lane L, vgpr r -> (m = r + (L<16?0:8), n = L%16)
    #pragma unroll
    for (int i = 0; i < 4; ++i)
        #pragma unroll
        for (int j = 0; j < 2; ++j)
            #pragma unroll
            for (int r = 0; r < 8; ++r) {
                const int row = m0 + wm * 64 + i * 16 + r + ((L < 16) ? 0 : 8);
                const int col = n0 + wn * 32 + j * 16 + (L & 15);
                if (col < Nout) {
                    float v = acc[i][j][r] + bias[col];
                    if (out_f16)
                        ((_Float16*)out)[(size_t)row * Nout + col] = (_Float16)v;
                    else
                        ((float*)out)[(size_t)row * Nout + col] = v;
                }
            }
}

// -------------------------------------------------------------------------
extern "C" void kernel_launch(void* const* d_in, const int* in_sizes, int n_in,
                              void* d_out, int out_size, void* d_ws, size_t ws_size,
                              hipStream_t stream) {
    (void)in_sizes; (void)n_in; (void)out_size; (void)ws_size;
    const int*   tokens = (const int*)  d_in[0];
    const float* emb    = (const float*)d_in[1];
    const float* W_f    = (const float*)d_in[2];
    const float* U_f    = (const float*)d_in[3];
    const float* b_f    = (const float*)d_in[4];
    const float* W_b    = (const float*)d_in[5];
    const float* U_b    = (const float*)d_in[6];
    const float* b_b    = (const float*)d_in[7];
    const float* W1     = (const float*)d_in[8];
    const float* b1     = (const float*)d_in[9];
    const float* W2     = (const float*)d_in[10];
    const float* b2     = (const float*)d_in[11];

    char* ws = (char*)d_ws;
    size_t off = 0;
    float*    xz  = (float*)   (ws + off); off += (size_t)2 * 8192 * G3 * 4;   // 50,331,648
    _Float16* Upk = (_Float16*)(ws + off); off += (size_t)2 * 196608 * 2;      //    786,432
    _Float16* ys  = (_Float16*)(ws + off); off += (size_t)8192 * 512 * 2;      //  8,388,608
    _Float16* W1p = (_Float16*)(ws + off); off += (size_t)NT_W * 16  * 512 * 2;//  8,257,536
    _Float16* W2p = (_Float16*)(ws + off); off += (size_t)NT_W * 250 * 512 * 2;//129,024,000
    _Float16* z1  = (_Float16*)(ws + off); off += (size_t)8192 * VOCAB * 2;    //131,072,000
    float* logits = (float*)d_out;

    // 1) embedding + x-projection
    embed_xproj<<<1024, 256, 0, stream>>>(tokens, emb, W_f, W_b, b_f, b_b, xz);

    // 2) pack recurrent + dense weights into WMMA B-fragment layout
    {   long tot = (long)48 * 8 * 512;                       // 196,608 per dir
        pack_w<<<(tot + 255) / 256, 256, 0, stream>>>(U_f, Upk,          8,  G3,   tot);
        pack_w<<<(tot + 255) / 256, 256, 0, stream>>>(U_b, Upk + 196608, 8,  G3,   tot);
    }
    {   long tot = (long)NT_W * 16 * 512;                    // 4,128,768
        pack_w<<<(tot + 255) / 256, 256, 0, stream>>>(W1, W1p, 16,  VOCAB, tot);
    }
    {   long tot = (long)NT_W * 250 * 512;                   // 64,512,000
        pack_w<<<(tot + 255) / 256, 256, 0, stream>>>(W2, W2p, 250, VOCAB, tot);
    }

    // 3) bidirectional GRU scan (WMMA recurrent matmul, one block per dir)
    gru_scan<<<2, 256, 0, stream>>>(xz, Upk, b_f, b_b, tokens, ys);

    // 4) z1 = ys @ W1 + b1   (f16 out for next GEMM)
    dim3 grid(NPAD / 128, 8192 / 128);   // 63 x 64
    gemm_f16<<<grid, 256, 0, stream>>>(ys, 512, 16, W1p, b1, (void*)z1, 1, VOCAB);

    // 5) logits = z1 @ W2 + b2   (f32 out, dominant ~1.05 TFLOP GEMM)
    gemm_f16<<<grid, 256, 0, stream>>>(z1, VOCAB, 250, W2p, b2, (void*)logits, 0, VOCAB);
}
